// ExternalKnowledge_12111807774827
// MI455X (gfx1250) — matrix operand order repaired and checked by reading, not compiled, not run on hardware
//
#include <hip/hip_runtime.h>
#include <hip/hip_bf16.h>

// Problem constants (from reference)
#define B_   32
#define M_   1024
#define T_   5
#define D_   256
#define L_   256
#define NCH  16      // m-chunks for the weighted-sum partial reduction

typedef __attribute__((ext_vector_type(16))) __bf16 v16bf;
typedef __attribute__((ext_vector_type(8)))  float  v8f;

union Frag16 { v16bf v; float4 f[2]; };          // 32 B: 16 bf16 per lane

__device__ __forceinline__ unsigned short f32_to_bf16_bits(float f) {
  unsigned int u = __float_as_uint(f);
  u += 0x7FFFu + ((u >> 16) & 1u);               // round-to-nearest-even
  return (unsigned short)(u >> 16);
}
__device__ __forceinline__ float bf16_bits_to_f32(unsigned short s) {
  return __uint_as_float(((unsigned int)s) << 16);
}

// ---------------------------------------------------------------------------
// 1) E[t][b][m][:] = sum_T C_t[story[b,m,:]] (+ dh row scatter) -> bf16 in ws.
//    One block per (b,m); 256 threads = D. Tables are L2-resident (131 MB).
// ---------------------------------------------------------------------------
__global__ __launch_bounds__(256)
void embed_bf16_kernel(const int* __restrict__ story,
                       const int* __restrict__ kb_len,
                       const int* __restrict__ conv_len,
                       const float* __restrict__ dh,       // [B][L][D]
                       const float* __restrict__ C1, const float* __restrict__ C2,
                       const float* __restrict__ C3, const float* __restrict__ C4,
                       unsigned short* __restrict__ E)     // [4][B][M][D] bf16
{
  const int bm = blockIdx.x;
  const int b  = bm >> 10;           // / M_
  const int m  = bm & (M_ - 1);
  const int d  = threadIdx.x;
  __shared__ int ids[T_];
  __shared__ int tpos;
  if (threadIdx.x < T_) ids[threadIdx.x] = story[bm * T_ + threadIdx.x];
  if (threadIdx.x == 0) {
    int t = m - (kb_len[b] - 1);
    tpos = (t >= 0 && t < conv_len[b]) ? t : -1;
  }
  __syncthreads();
  float lm = 0.f;
  if (tpos >= 0) lm = dh[((size_t)b * L_ + tpos) * D_ + d];
  const float* Cs[4] = {C1, C2, C3, C4};
  #pragma unroll
  for (int c = 0; c < 4; ++c) {
    float s = lm;
    #pragma unroll
    for (int t = 0; t < T_; ++t) s += Cs[c][(size_t)ids[t] * D_ + d];
    E[(((size_t)c * B_ + b) * M_ + m) * D_ + d] = f32_to_bf16_bits(s);
  }
}

// ---------------------------------------------------------------------------
// 1b) Tiny per-hop conversion of the hop vector to bf16 so the WMMA B-operand
//     is a pure b128 load (removes ~384 VALU cvt ops per tile from the hot
//     loop).  n = B*D elements.
// ---------------------------------------------------------------------------
__global__ __launch_bounds__(256)
void cvt_f32_to_bf16_kernel(const float* __restrict__ src,
                            unsigned short* __restrict__ dst)
{
  const int i = blockIdx.x * blockDim.x + threadIdx.x;
  dst[i] = f32_to_bf16_bits(src[i]);
}

// ---------------------------------------------------------------------------
// 2) Batched GEMV via WMMA: logits[b][m0..m0+15] = E[b] rows . u[b].
//    One wave per 16-row tile, 8x v_wmma_f32_16x16x32_bf16 over K=D.
//    B operand = bf16 u chunk replicated into all 16 columns (compute is free;
//    we are ~100x under the memory roofline). Optional post-scale by gp.
// ---------------------------------------------------------------------------
__global__ __launch_bounds__(256)
void logits_wmma_kernel(const unsigned short* __restrict__ E,  // [B][M][D] bf16
                        const unsigned short* __restrict__ ub, // [B][D] bf16
                        const float* __restrict__ scale,       // [B][M] or null
                        float* __restrict__ logits,            // [B][M]
                        int useScale)
{
  const int lane = threadIdx.x & 31;
  const int wave = threadIdx.x >> 5;
  const int tile = blockIdx.x * (blockDim.x >> 5) + wave;  // 0 .. B*M/16-1 exact
  const int b    = tile >> 6;                              // / (M_/16)
  const int m0   = (tile & 63) << 4;
  const unsigned short* Eb = E + ((size_t)b * M_ + m0) * D_;
  const unsigned short* uq = ub + (size_t)b * D_;
  const int rowA = lane & 15;
  const int hiA  = lane >> 4;                              // 0 or 1

  v8f c = {};
  #pragma unroll
  for (int k0 = 0; k0 < D_; k0 += 32) {
    // A: 16x32 bf16 tile, ISA lane layout (lanes<16: K 0..7 & 16..23;
    //    lanes>=16: K 8..15 & 24..31), two b128 loads per lane.
    Frag16 a;
    const unsigned short* rp = Eb + rowA * D_ + k0 + hiA * 8;
    a.f[0] = *(const float4*)(rp);
    a.f[1] = *(const float4*)(rp + 16);
    // B: 32x16, u chunk replicated across all 16 columns
    //    (lanes<16 hold K 0..7 & 16..23, lanes>=16 hold K 8..15 & 24..31).
    Frag16 bb;
    const unsigned short* up = uq + k0 + hiA * 8;
    bb.f[0] = *(const float4*)(up);
    bb.f[1] = *(const float4*)(up + 16);
    c = __builtin_amdgcn_wmma_f32_16x16x32_bf16(
        /*neg_a=*/false, a.v, /*neg_b=*/false, bb.v,
        /*c_mod=*/(short)0, c, /*reuse_a=*/false, /*reuse_b=*/false);
  }
  // Column N=0 of the 16x16 C tile: lane 0 holds M rows 0..7 in c[0..7],
  // lane 16 holds rows 8..15.
  if (rowA == 0) {
    const int mbase = m0 + hiA * 8;
    float* lp = logits + (size_t)b * M_ + mbase;
    #pragma unroll
    for (int r = 0; r < 8; ++r) {
      float x = c[r];
      if (useScale) x *= scale[(size_t)b * M_ + mbase + r];
      lp[r] = x;
    }
  }
}

// ---------------------------------------------------------------------------
// 3) Row softmax over M=1024, one block per b.
// ---------------------------------------------------------------------------
__global__ __launch_bounds__(256)
void softmax_kernel(const float* __restrict__ logits, float* __restrict__ p)
{
  __shared__ float red[256];
  const int b = blockIdx.x, t = threadIdx.x;
  const float* lb = logits + (size_t)b * M_;
  float v[4];
  float mx = -INFINITY;
  #pragma unroll
  for (int i = 0; i < 4; ++i) { v[i] = lb[t + i * 256]; mx = fmaxf(mx, v[i]); }
  red[t] = mx; __syncthreads();
  for (int s = 128; s > 0; s >>= 1) { if (t < s) red[t] = fmaxf(red[t], red[t + s]); __syncthreads(); }
  mx = red[0]; __syncthreads();
  float sum = 0.f;
  #pragma unroll
  for (int i = 0; i < 4; ++i) { v[i] = __expf(v[i] - mx); sum += v[i]; }
  red[t] = sum; __syncthreads();
  for (int s = 128; s > 0; s >>= 1) { if (t < s) red[t] += red[t + s]; __syncthreads(); }
  const float inv = 1.f / red[0];
  float* pb = p + (size_t)b * M_;
  #pragma unroll
  for (int i = 0; i < 4; ++i) pb[t + i * 256] = v[i] * inv;
}

// ---------------------------------------------------------------------------
// 4) u_new[d] = u[d] + sum_m w[m] * E[b][m][d], w = p (optionally * gp).
//    Split M into NCH chunks for occupancy; two-stage reduce.
// ---------------------------------------------------------------------------
__global__ __launch_bounds__(256)
void wsum_partial_kernel(const unsigned short* __restrict__ E,   // [B][M][D] bf16
                         const float* __restrict__ p,
                         const float* __restrict__ scale, int useScale,
                         float* __restrict__ partial)            // [B][NCH][D]
{
  const int b  = blockIdx.x / NCH;
  const int ch = blockIdx.x % NCH;
  const int d  = threadIdx.x;
  const int m0 = ch * (M_ / NCH);
  float acc = 0.f;
  for (int m = m0; m < m0 + M_ / NCH; ++m) {
    float w = p[(size_t)b * M_ + m];
    if (useScale) w *= scale[(size_t)b * M_ + m];
    acc += w * bf16_bits_to_f32(E[((size_t)b * M_ + m) * D_ + d]);
  }
  partial[((size_t)b * NCH + ch) * D_ + d] = acc;
}

__global__ __launch_bounds__(256)
void wsum_reduce_kernel(const float* __restrict__ partial,
                        const float* __restrict__ u_in,
                        float* __restrict__ u_out)
{
  const int b = blockIdx.x, d = threadIdx.x;
  float acc = u_in[(size_t)b * D_ + d];
  #pragma unroll
  for (int ch = 0; ch < NCH; ++ch) acc += partial[((size_t)b * NCH + ch) * D_ + d];
  u_out[(size_t)b * D_ + d] = acc;
}

__global__ __launch_bounds__(256)
void sigmoid_kernel(const float* __restrict__ logits, float* __restrict__ gp)
{
  const int i = blockIdx.x * blockDim.x + threadIdx.x;
  gp[i] = 1.f / (1.f + __expf(-logits[i]));
}

// ---------------------------------------------------------------------------
// Host orchestration. Workspace: E bf16 (64 MiB) + small f32/bf16 buffers.
// Pass 2 never materializes Eg: logits *= gp post-WMMA, weights *= gp pre-sum.
// ---------------------------------------------------------------------------
extern "C" void kernel_launch(void* const* d_in, const int* in_sizes, int n_in,
                              void* d_out, int out_size, void* d_ws, size_t ws_size,
                              hipStream_t stream)
{
  (void)in_sizes; (void)n_in; (void)out_size; (void)ws_size;
  const int*   story    = (const int*)  d_in[0];
  const int*   kb_len   = (const int*)  d_in[1];
  const int*   conv_len = (const int*)  d_in[2];
  const float* hidden   = (const float*)d_in[3];
  const float* dh       = (const float*)d_in[4];
  const float* query    = (const float*)d_in[5];
  const float* C1 = (const float*)d_in[6];
  const float* C2 = (const float*)d_in[7];
  const float* C3 = (const float*)d_in[8];
  const float* C4 = (const float*)d_in[9];

  char* ws = (char*)d_ws;
  unsigned short* E = (unsigned short*)ws;                       // 4*B*M*D bf16
  size_t off = (size_t)4 * B_ * M_ * D_ * sizeof(unsigned short);
  float* logits  = (float*)(ws + off); off += (size_t)B_ * M_ * sizeof(float);
  float* p       = (float*)(ws + off); off += (size_t)B_ * M_ * sizeof(float);
  float* partial = (float*)(ws + off); off += (size_t)B_ * NCH * D_ * sizeof(float);
  float* u0      = (float*)(ws + off); off += (size_t)B_ * D_ * sizeof(float);
  float* u1      = (float*)(ws + off); off += (size_t)B_ * D_ * sizeof(float);
  unsigned short* ub = (unsigned short*)(ws + off);              // [B][D] bf16
  off += (size_t)B_ * D_ * sizeof(unsigned short);

  float* out     = (float*)d_out;
  float* gp      = out;                        // [B,M]
  float* u_out   = gp + (size_t)B_ * M_;       // [B,D]
  float* qprob   = u_out + (size_t)B_ * D_;    // [B,M]
  float* qlogits = qprob + (size_t)B_ * M_;    // [B,M]

  const size_t EStride = (size_t)B_ * M_ * D_;
  const int logitsBlocks = (B_ * M_ / 16) / 8; // 8 waves (tiles) per 256-thr block
  const int cvtBlocks = (B_ * D_) / 256;       // 32 blocks

  embed_bf16_kernel<<<B_ * M_, 256, 0, stream>>>(story, kb_len, conv_len, dh,
                                                 C1, C2, C3, C4, E);

  // -------- pass 1: u = hidden, 3 hops, gp = sigmoid(last logits) --------
  hipMemcpyAsync(u0, hidden, (size_t)B_ * D_ * sizeof(float),
                 hipMemcpyDeviceToDevice, stream);
  float* ucur = u0; float* unext = u1;
  for (int i = 0; i < 3; ++i) {
    cvt_f32_to_bf16_kernel<<<cvtBlocks, 256, 0, stream>>>(ucur, ub);
    logits_wmma_kernel<<<logitsBlocks, 256, 0, stream>>>(E + i * EStride, ub,
                                                         nullptr, logits, 0);
    softmax_kernel<<<B_, 256, 0, stream>>>(logits, p);
    wsum_partial_kernel<<<B_ * NCH, 256, 0, stream>>>(E + (i + 1) * EStride, p,
                                                      nullptr, 0, partial);
    wsum_reduce_kernel<<<B_, 256, 0, stream>>>(partial, ucur, unext);
    float* t = ucur; ucur = unext; unext = t;
  }
  hipMemcpyAsync(u_out, ucur, (size_t)B_ * D_ * sizeof(float),
                 hipMemcpyDeviceToDevice, stream);
  sigmoid_kernel<<<(B_ * M_) / 256, 256, 0, stream>>>(logits, gp);

  // -------- pass 2: q = query, gp folded in; last hop emits qlogits/qprob ----
  float* qcur = unext; float* qnext = ucur;    // u already copied out; reuse
  hipMemcpyAsync(qcur, query, (size_t)B_ * D_ * sizeof(float),
                 hipMemcpyDeviceToDevice, stream);
  for (int i = 0; i < 2; ++i) {
    cvt_f32_to_bf16_kernel<<<cvtBlocks, 256, 0, stream>>>(qcur, ub);
    logits_wmma_kernel<<<logitsBlocks, 256, 0, stream>>>(E + i * EStride, ub,
                                                         gp, logits, 1);
    softmax_kernel<<<B_, 256, 0, stream>>>(logits, p);
    wsum_partial_kernel<<<B_ * NCH, 256, 0, stream>>>(E + (i + 1) * EStride, p,
                                                      gp, 1, partial);
    wsum_reduce_kernel<<<B_, 256, 0, stream>>>(partial, qcur, qnext);
    float* t = qcur; qcur = qnext; qnext = t;
  }
  cvt_f32_to_bf16_kernel<<<cvtBlocks, 256, 0, stream>>>(qcur, ub);
  logits_wmma_kernel<<<logitsBlocks, 256, 0, stream>>>(E + 2 * EStride, ub,
                                                       gp, qlogits, 1);
  softmax_kernel<<<B_, 256, 0, stream>>>(qlogits, qprob);
}
